// CustomTransformerEncoderLayer_55671366091113
// MI455X (gfx1250) — matrix operand order repaired and checked by reading, not compiled, not run on hardware
//
#include <hip/hip_runtime.h>
#include <math.h>

#define B_   2
#define S_   2048
#define D_   1024
#define H_   16
#define HD_  64
#define FF_  4096
#define MROWS (B_*S_)   // 4096

typedef _Float16 v16h __attribute__((ext_vector_type(16)));
typedef _Float16 v2h  __attribute__((ext_vector_type(2)));
typedef float    v8f  __attribute__((ext_vector_type(8)));
typedef int      gv4i __attribute__((vector_size(16)));   // matches builtin param

// ---- CDNA5 async global->LDS copy (ASYNCcnt path), with safe fallback ------
#if defined(__has_builtin)
#  if __has_builtin(__builtin_amdgcn_global_load_async_to_lds_b128)
#    define HAVE_ASYNC_LDS 1
#  endif
#endif
#ifndef HAVE_ASYNC_LDS
#  define HAVE_ASYNC_LDS 0
#endif

#if HAVE_ASYNC_LDS
static __device__ inline void async_copy_b128(const void* g, void* l) {
    __builtin_amdgcn_global_load_async_to_lds_b128(
        (gv4i*)g, (gv4i*)l, /*offset=*/0, /*cpol=*/0);
}
static __device__ inline void async_wait0() {
#  if __has_builtin(__builtin_amdgcn_s_wait_asynccnt)
    __builtin_amdgcn_s_wait_asynccnt(0);
#  else
    asm volatile("s_wait_asynccnt 0x0" ::: "memory");
#  endif
}
#endif

static __device__ inline v8f zero8() {
    v8f z;
#pragma unroll
    for (int i = 0; i < 8; ++i) z[i] = 0.0f;
    return z;
}

static __device__ inline v8f wmma32(v16h a, v16h b, v8f c) {
    // D = A(16x32 f16) * B(32x16 f16) + C(16x16 f32)
    return __builtin_amdgcn_wmma_f32_16x16x32_f16(
        /*neg_a=*/false, a, /*neg_b=*/false, b,
        /*c_mod=*/(short)0, c, /*reuse_a=*/false, /*reuse_b=*/false);
}

// A-matrix fragment (16x32 f16): M across lanes 0..15, K pairs with the
// lane-half interleave. Pairs {0..3,8..11}+4*half are contiguous 16B chunks,
// so this coalesces into 2x ds_load_b128 / global_load_b128.
static __device__ inline v16h load_a_frag(const _Float16* rowptr, int halfsel) {
    const v2h* p = (const v2h*)rowptr;
    const int base = halfsel * 4;
    v16h a;
#pragma unroll
    for (int j = 0; j < 8; ++j) {
        int idx = (j < 4 ? j : j + 4) + base;
        v2h t = p[idx];
        a[2 * j]     = t[0];
        a[2 * j + 1] = t[1];
    }
    return a;
}

// ---------------------------------------------------------------------------
// LayerNorm: f32 [rows][1024] -> f16 [rows][1024]
// ---------------------------------------------------------------------------
__global__ __launch_bounds__(256) void layernorm_kernel(
    const float* __restrict__ x, const float* __restrict__ g,
    const float* __restrict__ bta, _Float16* __restrict__ out) {
    const int row = blockIdx.x;
    const int t = threadIdx.x;
    const float* xr = x + (size_t)row * D_;
    float lv[4];
    float s = 0.f, s2 = 0.f;
#pragma unroll
    for (int i = 0; i < 4; ++i) {
        lv[i] = xr[t + i * 256];
        s  += lv[i];
        s2 += lv[i] * lv[i];
    }
#pragma unroll
    for (int off = 16; off >= 1; off >>= 1) {
        s  += __shfl_down(s, off, 32);
        s2 += __shfl_down(s2, off, 32);
    }
    __shared__ float red[16];
    __shared__ float stats[2];
    const int w = t >> 5, lane = t & 31;
    if (lane == 0) { red[w] = s; red[w + 8] = s2; }
    __syncthreads();
    if (t == 0) {
        float ts = 0.f, ts2 = 0.f;
        for (int i = 0; i < 8; ++i) { ts += red[i]; ts2 += red[i + 8]; }
        float mu  = ts * (1.0f / D_);
        float var = ts2 * (1.0f / D_) - mu * mu;
        stats[0] = mu;
        stats[1] = rsqrtf(var + 1e-5f);
    }
    __syncthreads();
    const float mu = stats[0], rstd = stats[1];
#pragma unroll
    for (int i = 0; i < 4; ++i) {
        int d = t + i * 256;
        out[(size_t)row * D_ + d] = (_Float16)((lv[i] - mu) * rstd * g[d] + bta[d]);
    }
}

// ---------------------------------------------------------------------------
// WMMA GEMM: C[M][N] = A(f16,[M][K]) * W(f32,[N][K])^T [+ bias] [+ resid]
// Block tile 128x128, 8 waves as 2(M)x4(N), wave tile 64x32 (4x2 subtiles),
// K-step 64 (two unrolled 32-deep WMMA halves). Compile-time epilogue flags.
// A tile staged via GLOBAL_LOAD_ASYNC_TO_LDS_B128 when available.
// ---------------------------------------------------------------------------
template <bool HAS_BIAS, bool HAS_RESID, bool OUT_F32>
__global__ __launch_bounds__(256) void gemm_wmma_kernel(
    const _Float16* __restrict__ A, const float* __restrict__ W,
    const float* __restrict__ bias, const float* __restrict__ resid,
    void* __restrict__ outp, int M, int N, int K) {
    __shared__ __align__(32) _Float16 As[128][64];   // activations [m][k]
    __shared__ __align__(32) _Float16 Bs[64][128];   // weights transposed [k][n]
    const int t = threadIdx.x;
    const int lane = t & 31;
    const int w = t >> 5;
    const int wm = w & 1, wn = w >> 1;               // 2 x 4 wave grid
    const int bm = blockIdx.y * 128, bn = blockIdx.x * 128;
    const int r = lane & 15, hsel = lane >> 4;

    v8f acc[4][2];
#pragma unroll
    for (int i = 0; i < 4; ++i)
#pragma unroll
        for (int j = 0; j < 2; ++j) acc[i][j] = zero8();

    const int ar = t >> 1, ac = (t & 1) * 32;        // A: 2 thr/row, 32 f16 each
    const int wr = t >> 1, wc = (t & 1) * 32;        // W: 2 thr/row, 32 f32 each

    for (int k0 = 0; k0 < K; k0 += 64) {
        if (k0 + 64 < K) {
            __builtin_prefetch(A + (size_t)(bm + ar) * K + k0 + 64 + ac, 0, 2);
            __builtin_prefetch(W + (size_t)(bn + wr) * K + k0 + 64 + wc, 0, 2);
        }
        __syncthreads();
        {
            const _Float16* agp = A + (size_t)(bm + ar) * K + k0 + ac;
#if HAVE_ASYNC_LDS
#pragma unroll
            for (int i = 0; i < 4; ++i)
                async_copy_b128(agp + 8 * i, &As[ar][ac + 8 * i]);
#else
            const uint4* ap = (const uint4*)agp;
#pragma unroll
            for (int i = 0; i < 4; ++i)
                *(uint4*)&As[ar][ac + 8 * i] = ap[i];
#endif
        }
        {
            const float4* wp = (const float4*)(W + (size_t)(bn + wr) * K + k0 + wc);
#pragma unroll
            for (int i = 0; i < 8; ++i) {
                float4 f = wp[i];
                Bs[wc + 4 * i + 0][wr] = (_Float16)f.x;
                Bs[wc + 4 * i + 1][wr] = (_Float16)f.y;
                Bs[wc + 4 * i + 2][wr] = (_Float16)f.z;
                Bs[wc + 4 * i + 3][wr] = (_Float16)f.w;
            }
        }
#if HAVE_ASYNC_LDS
        async_wait0();
#endif
        __syncthreads();

#pragma unroll
        for (int kh = 0; kh < 2; ++kh) {
            v16h af[4], bf[2];
#pragma unroll
            for (int i = 0; i < 4; ++i)
                af[i] = load_a_frag(&As[wm * 64 + i * 16 + r][kh * 32], hsel);
#pragma unroll
            for (int j = 0; j < 2; ++j)
#pragma unroll
                for (int s = 0; s < 16; ++s)
                    bf[j][s] = Bs[kh * 32 + (s < 8 ? s : s + 8) + 8 * hsel][wn * 32 + j * 16 + r];
#pragma unroll
            for (int i = 0; i < 4; ++i)
#pragma unroll
                for (int j = 0; j < 2; ++j)
                    acc[i][j] = wmma32(af[i], bf[j], acc[i][j]);
        }
    }

#pragma unroll
    for (int i = 0; i < 4; ++i)
#pragma unroll
        for (int j = 0; j < 2; ++j) {
            int col = bn + wn * 32 + j * 16 + r;
            float bv = HAS_BIAS ? bias[col] : 0.0f;
            int rowb = bm + wm * 64 + i * 16 + hsel * 8;
#pragma unroll
            for (int r8 = 0; r8 < 8; ++r8) {
                int row = rowb + r8;
                float val = acc[i][j][r8] + bv;
                if (HAS_RESID) val += resid[(size_t)row * N + col];
                if (OUT_F32) ((float*)outp)[(size_t)row * N + col] = val;
                else         ((_Float16*)outp)[(size_t)row * N + col] = (_Float16)val;
            }
        }
}

// ---------------------------------------------------------------------------
// RoPE on full d_model (before head split), applied to q and k in place (f16)
// ---------------------------------------------------------------------------
__global__ __launch_bounds__(256) void rope_kernel(
    _Float16* __restrict__ q, _Float16* __restrict__ k,
    const float* __restrict__ cosp, const float* __restrict__ sinp) {
    const int row = blockIdx.x;          // 0..MROWS-1
    const int s = row & (S_ - 1);
    const size_t base = (size_t)row * D_;
#pragma unroll
    for (int i = 0; i < 2; ++i) {
        int d = threadIdx.x + i * 256;   // d in [0, 512)
        float c1 = cosp[(size_t)s * D_ + d];
        float s1 = sinp[(size_t)s * D_ + d];
        float c2 = cosp[(size_t)s * D_ + d + 512];
        float s2 = sinp[(size_t)s * D_ + d + 512];
        float q1 = (float)q[base + d], q2 = (float)q[base + d + 512];
        q[base + d]       = (_Float16)(q1 * c1 - q2 * s1);
        q[base + d + 512] = (_Float16)(q2 * c2 + q1 * s2);
        float k1 = (float)k[base + d], k2 = (float)k[base + d + 512];
        k[base + d]       = (_Float16)(k1 * c1 - k2 * s1);
        k[base + d + 512] = (_Float16)(k2 * c2 + k1 * s2);
    }
}

// ---------------------------------------------------------------------------
// Flash attention: grid (B*H, S/128); 8 waves each own 16 q-rows.
// Per 64-key step: stage K (64x64) and V (64x64) in LDS (V via async-LDS),
// 8 WMMA for scores, online softmax, P via per-wave LDS -> A fragments,
// 8 WMMA for P*V.
// ---------------------------------------------------------------------------
__global__ __launch_bounds__(256) void attn_kernel(
    const _Float16* __restrict__ q, const _Float16* __restrict__ k,
    const _Float16* __restrict__ v, const unsigned char* __restrict__ mask,
    _Float16* __restrict__ out) {
    __shared__ __align__(32) _Float16 Ks[64][64];     // [key][feature]
    __shared__ __align__(32) _Float16 Vs[64][64];     // [key][feature]
    __shared__ __align__(32) _Float16 Ps[8][16][64];  // per-wave P tile
    const int b = blockIdx.x >> 4;
    const int h = blockIdx.x & 15;
    const int t = threadIdx.x, lane = t & 31, w = t >> 5;
    const int r = lane & 15, hsel = lane >> 4;
    const int q0 = blockIdx.y * 128 + w * 16;

    v16h qf0, qf1;
    {
        const _Float16* qrow = q + (size_t)(b * S_ + q0 + r) * D_ + h * HD_;
        qf0 = load_a_frag(qrow, hsel);        // features 0..31
        qf1 = load_a_frag(qrow + 32, hsel);   // features 32..63
    }
    float rmax[8], rsum[8];
    v8f o[4];
#pragma unroll
    for (int i = 0; i < 8; ++i) { rmax[i] = -1e30f; rsum[i] = 0.f; }
#pragma unroll
    for (int n = 0; n < 4; ++n) o[n] = zero8();

    const int kk = t >> 2, f0 = (t & 3) * 16;   // staging: 4 thr/key, 16 f16 each
    const float scale = 0.125f;                 // HD^-0.5

    for (int key0 = 0; key0 < S_; key0 += 64) {
        __syncthreads();
        {
            const _Float16* vg = v + (size_t)(b * S_ + key0 + kk) * D_ + h * HD_ + f0;
#if HAVE_ASYNC_LDS
            async_copy_b128(vg,     &Vs[kk][f0]);
            async_copy_b128(vg + 8, &Vs[kk][f0 + 8]);
#else
            const uint4* vp = (const uint4*)vg;
            *(uint4*)&Vs[kk][f0]     = vp[0];
            *(uint4*)&Vs[kk][f0 + 8] = vp[1];
#endif
            const uint4* kp = (const uint4*)(k + (size_t)(b * S_ + key0 + kk) * D_ + h * HD_ + f0);
            uint4 kd0 = kp[0], kd1 = kp[1];
            _Float16 tmp[16];
            *(uint4*)&tmp[0] = kd0;
            *(uint4*)&tmp[8] = kd1;
#pragma unroll
            for (int i = 0; i < 16; ++i) Ks[kk][f0 + i] = tmp[i];  // [key][feature] store
        }
#if HAVE_ASYNC_LDS
        async_wait0();
#endif
        __syncthreads();

        v8f sacc[4];
#pragma unroll
        for (int j = 0; j < 4; ++j) sacc[j] = zero8();
#pragma unroll
        for (int j = 0; j < 4; ++j) {
            v16h kfa = load_a_frag(&Ks[j * 16 + r][0], hsel);   // features 0..31
            v16h kfb = load_a_frag(&Ks[j * 16 + r][32], hsel);  // features 32..63
            sacc[j] = wmma32(qf0, kfa, sacc[j]);
            sacc[j] = wmma32(qf1, kfb, sacc[j]);
        }
        // scale + padding mask
        unsigned char mj[4];
#pragma unroll
        for (int j = 0; j < 4; ++j) mj[j] = mask[(size_t)b * S_ + key0 + j * 16 + r];
#pragma unroll
        for (int j = 0; j < 4; ++j)
#pragma unroll
            for (int r8 = 0; r8 < 8; ++r8) {
                float sv = sacc[j][r8] * scale;
                if (mj[j]) sv = -1e30f;
                sacc[j][r8] = sv;
            }
        // online softmax (row stats in 16-lane groups; xor<=8 stays in-group)
#pragma unroll
        for (int r8 = 0; r8 < 8; ++r8) {
            float m = fmaxf(fmaxf(sacc[0][r8], sacc[1][r8]),
                            fmaxf(sacc[2][r8], sacc[3][r8]));
#pragma unroll
            for (int off = 8; off >= 1; off >>= 1) m = fmaxf(m, __shfl_xor(m, off, 32));
            float nm = fmaxf(rmax[r8], m);
            float corr = __expf(rmax[r8] - nm);
            float ps = 0.f;
#pragma unroll
            for (int j = 0; j < 4; ++j) {
                float p = __expf(sacc[j][r8] - nm);
                sacc[j][r8] = p;
                ps += p;
            }
#pragma unroll
            for (int off = 8; off >= 1; off >>= 1) ps += __shfl_xor(ps, off, 32);
            rsum[r8] = rsum[r8] * corr + ps;
            rmax[r8] = nm;
#pragma unroll
            for (int n = 0; n < 4; ++n) o[n][r8] *= corr;
        }
        // P (C layout) -> per-wave LDS -> A fragment layout (same-wave DS in-order)
#pragma unroll
        for (int r8 = 0; r8 < 8; ++r8) {
            int prow = r8 + 8 * hsel;
#pragma unroll
            for (int j = 0; j < 4; ++j)
                Ps[w][prow][r + 16 * j] = (_Float16)sacc[j][r8];
        }
        v16h pf0 = load_a_frag(&Ps[w][r][0],  hsel);   // keys 0..31
        v16h pf1 = load_a_frag(&Ps[w][r][32], hsel);   // keys 32..63
#pragma unroll
        for (int n = 0; n < 4; ++n) {
            v16h vfa, vfb;
#pragma unroll
            for (int s = 0; s < 16; ++s) {
                int kx = (s < 8 ? s : s + 8) + 8 * hsel;
                vfa[s] = Vs[kx][n * 16 + r];
                vfb[s] = Vs[32 + kx][n * 16 + r];
            }
            o[n] = wmma32(pf0, vfa, o[n]);
            o[n] = wmma32(pf1, vfb, o[n]);
        }
    }
    // normalize and store (heads re-interleaved into [row][h*64 + d])
    float rinv[8];
#pragma unroll
    for (int r8 = 0; r8 < 8; ++r8) rinv[r8] = 1.0f / rsum[r8];
#pragma unroll
    for (int n = 0; n < 4; ++n)
#pragma unroll
        for (int r8 = 0; r8 < 8; ++r8) {
            int row = q0 + r8 + 8 * hsel;
            out[(size_t)(b * S_ + row) * D_ + h * HD_ + n * 16 + r] =
                (_Float16)(o[n][r8] * rinv[r8]);
        }
}

// ---------------------------------------------------------------------------
// h = silu(gate) * up, written in-place into gate (f16)
// ---------------------------------------------------------------------------
__global__ __launch_bounds__(256) void silu_mul_kernel(
    _Float16* __restrict__ g, const _Float16* __restrict__ u) {
    size_t i = ((size_t)blockIdx.x * 256 + threadIdx.x) * 4;
#pragma unroll
    for (int j = 0; j < 4; ++j) {
        float gv = (float)g[i + j];
        float uv = (float)u[i + j];
        float s = gv / (1.0f + __expf(-gv));
        g[i + j] = (_Float16)(s * uv);
    }
}

// ---------------------------------------------------------------------------
extern "C" void kernel_launch(void* const* d_in, const int* in_sizes, int n_in,
                              void* d_out, int out_size, void* d_ws, size_t ws_size,
                              hipStream_t stream) {
    const float* src  = (const float*)d_in[0];
    const float* cosp = (const float*)d_in[1];
    const float* sinp = (const float*)d_in[2];
    const unsigned char* mask = (const unsigned char*)d_in[3];
    const float* Wq = (const float*)d_in[4];
    const float* bq = (const float*)d_in[5];
    const float* Wk = (const float*)d_in[6];
    const float* bk = (const float*)d_in[7];
    const float* Wv = (const float*)d_in[8];
    const float* bv = (const float*)d_in[9];
    const float* Wo = (const float*)d_in[10];
    const float* bo = (const float*)d_in[11];
    const float* g1 = (const float*)d_in[12];
    const float* b1 = (const float*)d_in[13];
    const float* g2 = (const float*)d_in[14];
    const float* b2 = (const float*)d_in[15];
    const float* W1 = (const float*)d_in[16];
    const float* W3 = (const float*)d_in[17];
    const float* W2 = (const float*)d_in[18];

    char* ws = (char*)d_ws;
    const size_t MB = 1ull << 20;
    _Float16* xh   = (_Float16*)(ws + 0 * MB);    //  8 MB  LN1 output
    _Float16* qb   = (_Float16*)(ws + 8 * MB);    //  8 MB  q (f16)
    _Float16* kb   = (_Float16*)(ws + 16 * MB);   //  8 MB  k (f16)
    _Float16* vb   = (_Float16*)(ws + 24 * MB);   //  8 MB  v (f16)
    _Float16* attn = (_Float16*)(ws + 32 * MB);   //  8 MB  attention output
    float*    src2 = (float*)   (ws + 40 * MB);   // 16 MB  residual stream 2
    _Float16* x2h  = (_Float16*)(ws + 56 * MB);   //  8 MB  LN2 output
    _Float16* gate = (_Float16*)(ws + 64 * MB);   // 32 MB  gate / fused h
    _Float16* up   = (_Float16*)(ws + 96 * MB);   // 32 MB  up

    dim3 blk(256);
    dim3 gD(D_ / 128, MROWS / 128);   // (8, 32)
    dim3 gF(FF_ / 128, MROWS / 128);  // (32, 32)

    layernorm_kernel<<<MROWS, blk, 0, stream>>>(src, g1, b1, xh);
    gemm_wmma_kernel<true,  false, false><<<gD, blk, 0, stream>>>(xh, Wq, bq, nullptr, qb, MROWS, D_, D_);
    gemm_wmma_kernel<true,  false, false><<<gD, blk, 0, stream>>>(xh, Wk, bk, nullptr, kb, MROWS, D_, D_);
    gemm_wmma_kernel<true,  false, false><<<gD, blk, 0, stream>>>(xh, Wv, bv, nullptr, vb, MROWS, D_, D_);
    rope_kernel<<<MROWS, blk, 0, stream>>>(qb, kb, cosp, sinp);
    attn_kernel<<<dim3(B_ * H_, S_ / 128), blk, 0, stream>>>(qb, kb, vb, mask, attn);
    gemm_wmma_kernel<true,  true,  true ><<<gD, blk, 0, stream>>>(attn, Wo, bo, src, src2, MROWS, D_, D_);
    layernorm_kernel<<<MROWS, blk, 0, stream>>>(src2, g2, b2, x2h);
    gemm_wmma_kernel<false, false, false><<<gF, blk, 0, stream>>>(x2h, W1, nullptr, nullptr, gate, MROWS, FF_, D_);
    gemm_wmma_kernel<false, false, false><<<gF, blk, 0, stream>>>(x2h, W3, nullptr, nullptr, up, MROWS, FF_, D_);
    silu_mul_kernel<<<(unsigned)((size_t)MROWS * FF_ / 1024), blk, 0, stream>>>(gate, up);
    gemm_wmma_kernel<false, true,  true ><<<gD, blk, 0, stream>>>(gate, W2, nullptr, src2, (float*)d_out, MROWS, D_, FF_);

    (void)in_sizes; (void)n_in; (void)out_size; (void)ws_size;
}
